// GINE_2D_12352325943372
// MI455X (gfx1250) — matrix operand order repaired and checked by reading, not compiled, not run on hardware
//
#include <hip/hip_runtime.h>

// ---------------------------------------------------------------------------
// CDNA5 (gfx1250) wave32 WMMA types
// ---------------------------------------------------------------------------
typedef __bf16 v16bf __attribute__((ext_vector_type(16)));
typedef float  v8f   __attribute__((ext_vector_type(8)));

#define DH 128
#define DE 16

__device__ __forceinline__ void pack4(v16bf& d, int base, float4 f) {
    d[base + 0] = (__bf16)f.x;
    d[base + 1] = (__bf16)f.y;
    d[base + 2] = (__bf16)f.z;
    d[base + 3] = (__bf16)f.w;
}

__device__ __forceinline__ int imin(int a, int b) { return a < b ? a : b; }

// ---------------------------------------------------------------------------
// utility kernels
// ---------------------------------------------------------------------------
__global__ void zero_kernel(float* __restrict__ p, long long n) {
    long long i = (long long)blockIdx.x * blockDim.x + threadIdx.x;
    long long stride = (long long)gridDim.x * blockDim.x;
    for (; i < n; i += stride) p[i] = 0.0f;
}

__global__ void count_kernel(const long long* __restrict__ batch,
                             float* __restrict__ cnt, int n) {
    int i = blockIdx.x * blockDim.x + threadIdx.x;
    if (i < n) atomicAdd(&cnt[(int)batch[i]], 1.0f);
}

// f32 -> bf16 weight conversion (one-time prep, weights are tiny)
__global__ void cvt_bf16_kernel(const float* __restrict__ src,
                                __bf16* __restrict__ dst, int n) {
    int i = blockIdx.x * blockDim.x + threadIdx.x;
    if (i < n) dst[i] = (__bf16)src[i];
}

// edge weight [rows,16] f32 -> [rows,32] bf16 zero-padded along K
__global__ void pad_ew_kernel(const float* __restrict__ src,
                              __bf16* __restrict__ dst, int rows) {
    int i = blockIdx.x * blockDim.x + threadIdx.x;
    if (i < rows * 32) {
        int r = i >> 5, k = i & 31;
        dst[i] = (k < DE) ? (__bf16)src[r * DE + k] : (__bf16)0.0f;
    }
}

// ---------------------------------------------------------------------------
// Edge message kernel: per wave, 16 edges. Branchless.
//   lin  = edge_attr[e] @ ew^T + eb   (WMMA bf16, K=16 zero-padded to 32)
//   msg  = relu(xfeat[src[e]] + lin) ; atomicAdd(aggr[dst[e]], msg)
// OOB rows: loads are index-clamped (garbage only feeds discarded C rows),
// scatters are redirected to dummy row N of aggr (never read).
// aggr (<= 51 MB) is L2-resident so atomics stay on-chip.
// ---------------------------------------------------------------------------
template <int DF>
__global__ void edge_msg_kernel(const float* __restrict__ xfeat,      // [N, DF]
                                const float* __restrict__ eattr,      // [E, 16]
                                const long long* __restrict__ srcI,   // [E]
                                const long long* __restrict__ dstI,   // [E]
                                const __bf16* __restrict__ ewp,       // [DF, 32]
                                const float* __restrict__ ebias,      // [DF]
                                float* __restrict__ aggr,             // [N+1, DF]
                                int E, int dummyOff) {
    const int lane = threadIdx.x & 31;
    const int wave = threadIdx.x >> 5;
    const int half = lane >> 4;
    const int nloc = lane & 15;
    const int ebase = (blockIdx.x * (blockDim.x >> 5) + wave) * 16;
    if (ebase >= E) return;                       // wave-uniform

    constexpr int NT = DF / 16;

    // ---- B fragments: preload all weight tiles (bf16, K runs half*8 and 16+half*8)
    v16bf b[NT];
    #pragma unroll
    for (int t = 0; t < NT; ++t) {
        const __bf16* wq = ewp + (size_t)(t * 16 + nloc) * 32 + half * 8;
        #pragma unroll
        for (int e = 0; e < 8; ++e) { b[t][e] = wq[e]; b[t][8 + e] = wq[16 + e]; }
    }

    // ---- A fragment: rows = 16 edges (clamped), K = 16 (elems 8..15 -> 0)
    v16bf a;
    {
        const int rowc = imin(ebase + nloc, E - 1);
        const float* ep = eattr + (size_t)rowc * DE + half * 8;
        pack4(a, 0, *(const float4*)(ep));
        pack4(a, 4, *(const float4*)(ep + 4));
        #pragma unroll
        for (int e = 8; e < 16; ++e) a[e] = (__bf16)0.0f;
    }

    // ---- 32-bit element offsets: gather (clamped) and scatter (dummy if OOB)
    int soff[8], doff[8];
    #pragma unroll
    for (int r = 0; r < 8; ++r) {
        int erow = ebase + half * 8 + r;
        int erc  = imin(erow, E - 1);
        soff[r] = (int)srcI[erc] * DF;
        doff[r] = (erow < E) ? (int)dstI[erc] * DF : dummyOff;
    }

    #pragma unroll
    for (int t = 0; t < NT; ++t) {
        const int o = t * 16 + nloc;
        const float bv = ebias[o];
        v8f acc;
        #pragma unroll
        for (int i = 0; i < 8; ++i) acc[i] = bv;   // C = bias broadcast per column
        acc = __builtin_amdgcn_wmma_f32_16x16x32_bf16(
            false, a, false, b[t], (short)0, acc, false, false);

        float xv[8];
        #pragma unroll
        for (int r = 0; r < 8; ++r) xv[r] = xfeat[soff[r] + o];  // load clause
        #pragma unroll
        for (int r = 0; r < 8; ++r) {
            float v = fmaxf(acc[r] + xv[r], 0.0f);
            atomicAdd(&aggr[doff[r] + o], v);
        }
    }
}

// ---------------------------------------------------------------------------
// Node GEMM: out[n, 0:128] = epi( (inA[n] (+ inB[n])) @ W^T + bias ), branchless.
//   KD  : input width (64 or 128); Wbf is bf16 [128, KD]
//   EPI : 0 -> dense relu store (OOB rows -> dummy row N)
//         1 -> fused pooled atomicAdd by batch[] (OOB rows -> dummy row G)
// ---------------------------------------------------------------------------
template <int KD, int EPI>
__global__ void node_gemm_kernel(const float* __restrict__ inA,   // [N, KD]
                                 const float* __restrict__ inB,   // [N, KD] | null
                                 const __bf16* __restrict__ Wbf,  // [128, KD]
                                 const float* __restrict__ bias,  // [128]
                                 const long long* __restrict__ batch,
                                 float* __restrict__ outp,  // [N+1,128]|[G+1,128]
                                 int N, int dummyOff) {
    const int lane = threadIdx.x & 31;
    const int wave = threadIdx.x >> 5;
    const int half = lane >> 4;
    const int nloc = lane & 15;
    const int nb = (blockIdx.x * (blockDim.x >> 5) + wave) * 16;
    if (nb >= N) return;                          // wave-uniform

    constexpr int KC = KD / 32;
    v16bf a[KC];
    {
        const int rowc = imin(nb + nloc, N - 1);
        const float* pa = inA + (size_t)rowc * KD;
        const float* pb = inB ? (inB + (size_t)rowc * KD) : nullptr;
        #pragma unroll
        for (int kc = 0; kc < KC; ++kc) {
            const float* q0 = pa + kc * 32 + half * 16;      // K run 16h..16h+7
            const float* q1 = pa + kc * 32 + half * 16 + 8;  // K run 16h+8..16h+15
            float4 f0 = *(const float4*)(q0);
            float4 f1 = *(const float4*)(q0 + 4);
            float4 f2 = *(const float4*)(q1);
            float4 f3 = *(const float4*)(q1 + 4);
            if (pb) {
                const float* r0 = pb + kc * 32 + half * 16;
                const float* r1 = pb + kc * 32 + half * 16 + 8;
                float4 g0 = *(const float4*)(r0);
                float4 g1 = *(const float4*)(r0 + 4);
                float4 g2 = *(const float4*)(r1);
                float4 g3 = *(const float4*)(r1 + 4);
                f0.x += g0.x; f0.y += g0.y; f0.z += g0.z; f0.w += g0.w;
                f1.x += g1.x; f1.y += g1.y; f1.z += g1.z; f1.w += g1.w;
                f2.x += g2.x; f2.y += g2.y; f2.z += g2.z; f2.w += g2.w;
                f3.x += g3.x; f3.y += g3.y; f3.z += g3.z; f3.w += g3.w;
            }
            pack4(a[kc], 0,  f0);
            pack4(a[kc], 4,  f1);
            pack4(a[kc], 8,  f2);
            pack4(a[kc], 12, f3);
        }
    }

    // 32-bit output element offsets per C-fragment row (dummy row if OOB)
    int ooff[8];
    #pragma unroll
    for (int r = 0; r < 8; ++r) {
        int node = nb + half * 8 + r;
        int nc   = imin(node, N - 1);
        int base = (EPI == 1) ? (int)batch[nc] * DH : nc * DH;
        ooff[r] = (node < N) ? base : dummyOff;
    }

    #pragma unroll
    for (int t = 0; t < 8; ++t) {
        const int o = t * 16 + nloc;
        const float bv = bias[o];
        v8f acc;
        #pragma unroll
        for (int i = 0; i < 8; ++i) acc[i] = bv;   // C = bias broadcast
        #pragma unroll
        for (int kc = 0; kc < KC; ++kc) {
            v16bf b = *(const v16bf*)(Wbf + (size_t)o * KD + kc * 32 + half * 16);
            acc = __builtin_amdgcn_wmma_f32_16x16x32_bf16(
                false, a[kc], false, b, (short)0, acc, false, false);
        }
        #pragma unroll
        for (int r = 0; r < 8; ++r) {
            float v = fmaxf(acc[r], 0.0f);
            if (EPI == 0) outp[ooff[r] + o] = v;
            else          atomicAdd(&outp[ooff[r] + o], v);
        }
    }
}

// ---------------------------------------------------------------------------
// Final: out[g] = dot(pool[g], wf) / max(cnt[g],1) + bf
// ---------------------------------------------------------------------------
__global__ void final_kernel(const float* __restrict__ pool,  // [G,128]
                             const float* __restrict__ cnt,   // [G]
                             const float* __restrict__ wf,    // [128]
                             const float* __restrict__ bfp,   // [1]
                             float* __restrict__ out) {
    __shared__ float red[DH];
    const int g = blockIdx.x;
    const int o = threadIdx.x;
    red[o] = pool[(size_t)g * DH + o] * wf[o];
    __syncthreads();
    for (int s = DH / 2; s > 0; s >>= 1) {
        if (o < s) red[o] += red[o + s];
        __syncthreads();
    }
    if (o == 0) out[g] = red[0] / fmaxf(cnt[g], 1.0f) + bfp[0];
}

// ---------------------------------------------------------------------------
// launch
// ---------------------------------------------------------------------------
extern "C" void kernel_launch(void* const* d_in, const int* in_sizes, int n_in,
                              void* d_out, int out_size, void* d_ws, size_t ws_size,
                              hipStream_t stream) {
    const float*     x     = (const float*)d_in[0];
    const long long* ei    = (const long long*)d_in[1];   // [2, E] int64
    const float*     eattr = (const float*)d_in[2];
    const long long* batch = (const long long*)d_in[3];
    const float* ew1 = (const float*)d_in[4],  *eb1 = (const float*)d_in[5];
    const float* w1a = (const float*)d_in[6],  *b1a = (const float*)d_in[7];
    const float* w1b = (const float*)d_in[8],  *b1b = (const float*)d_in[9];
    const float* ew2 = (const float*)d_in[10], *eb2 = (const float*)d_in[11];
    const float* w2a = (const float*)d_in[12], *b2a = (const float*)d_in[13];
    const float* w2b = (const float*)d_in[14], *b2b = (const float*)d_in[15];
    const float* wf  = (const float*)d_in[16], *bfp = (const float*)d_in[17];
    float* out = (float*)d_out;

    const int N = in_sizes[0] / 64;   // 100000
    const int E = in_sizes[2] / DE;   // 1600000
    const int G = out_size;           // 256 (DOUT = 1)

    const long long* srcI = ei;
    const long long* dstI = ei + E;

    // ws layout (floats), each N-row buffer has +1 dummy row for OOB redirect:
    // AG[(N+1)*128] | TMP[(N+1)*128] | H1[(N+1)*128] | POOL[(G+1)*128] | CNT[G]
    // then 64B-aligned bf16 weight copies.
    float* AG   = (float*)d_ws;
    float* TMP  = AG  + (size_t)(N + 1) * DH;
    float* H1   = TMP + (size_t)(N + 1) * DH;
    float* POOL = H1  + (size_t)(N + 1) * DH;
    float* CNT  = POOL + (size_t)(G + 1) * DH;
    uintptr_t wp = ((uintptr_t)(CNT + G) + 63) & ~(uintptr_t)63;
    __bf16* w1a_bf = (__bf16*)wp;               // 128*64
    __bf16* w1b_bf = w1a_bf + 128 * 64;         // 128*128
    __bf16* w2a_bf = w1b_bf + 128 * 128;        // 128*128
    __bf16* w2b_bf = w2a_bf + 128 * 128;        // 128*128
    __bf16* ew1p   = w2b_bf + 128 * 128;        // 64*32
    __bf16* ew2p   = ew1p   + 64 * 32;          // 128*32

    const int edgeBlocks = (E + 127) / 128;   // 8 waves x 16 edges per block
    const int nodeBlocks = (N + 127) / 128;   // 8 waves x 16 nodes per block

    // ---- weight prep (tiny) ----
    cvt_bf16_kernel<<<(128 * 64  + 255) / 256, 256, 0, stream>>>(w1a, w1a_bf, 128 * 64);
    cvt_bf16_kernel<<<(128 * 128 + 255) / 256, 256, 0, stream>>>(w1b, w1b_bf, 128 * 128);
    cvt_bf16_kernel<<<(128 * 128 + 255) / 256, 256, 0, stream>>>(w2a, w2a_bf, 128 * 128);
    cvt_bf16_kernel<<<(128 * 128 + 255) / 256, 256, 0, stream>>>(w2b, w2b_bf, 128 * 128);
    pad_ew_kernel<<<(64 * 32  + 255) / 256, 256, 0, stream>>>(ew1, ew1p, 64);
    pad_ew_kernel<<<(128 * 32 + 255) / 256, 256, 0, stream>>>(ew2, ew2p, 128);

    // ---- Layer 1 ----
    zero_kernel<<<4096, 256, 0, stream>>>(AG, (long long)N * 64);
    edge_msg_kernel<64><<<edgeBlocks, 256, 0, stream>>>(
        x, eattr, srcI, dstI, ew1p, eb1, AG, E, N * 64);
    node_gemm_kernel<64, 0><<<nodeBlocks, 256, 0, stream>>>(
        x, AG, w1a_bf, b1a, nullptr, TMP, N, N * DH);
    node_gemm_kernel<128, 0><<<nodeBlocks, 256, 0, stream>>>(
        TMP, nullptr, w1b_bf, b1b, nullptr, H1, N, N * DH);

    // ---- Layer 2 ----
    zero_kernel<<<4096, 256, 0, stream>>>(AG, (long long)N * DH);
    edge_msg_kernel<128><<<edgeBlocks, 256, 0, stream>>>(
        H1, eattr, srcI, dstI, ew2p, eb2, AG, E, N * DH);
    node_gemm_kernel<128, 0><<<nodeBlocks, 256, 0, stream>>>(
        H1, AG, w2a_bf, b2a, nullptr, TMP, N, N * DH);

    // ---- Pool + head ----
    zero_kernel<<<64, 256, 0, stream>>>(POOL, (long long)(G + 1) * DH + G);
    count_kernel<<<(N + 255) / 256, 256, 0, stream>>>(batch, CNT, N);
    node_gemm_kernel<128, 1><<<nodeBlocks, 256, 0, stream>>>(
        TMP, nullptr, w2b_bf, b2b, batch, POOL, N, G * DH);
    final_kernel<<<G, DH, 0, stream>>>(POOL, CNT, wf, bfp, out);
}
